// CausalAttentionLayer_82231443849521
// MI455X (gfx1250) — matrix operand order repaired and checked by reading, not compiled, not run on hardware
//
#include <hip/hip_runtime.h>

// MI455X (gfx1250) flash-attention for pixelSNAIL causal layer:
//   softmax over FULL row, strict causal mask applied to numerator only.
// Pass 1: K f32 -> bf16.  Pass 2: V f32 -> bf16 transposed [B][CM][N].
// Pass 3: flash kernel, bf16 WMMA, f32 accum, async double-buffered LDS tiles.

typedef __attribute__((ext_vector_type(16))) __bf16 v16bf;
typedef __attribute__((ext_vector_type(8)))  __bf16 v8bf;
typedef __attribute__((ext_vector_type(4)))  __bf16 v4bf;
typedef __attribute__((ext_vector_type(8)))  float  v8f;
typedef __attribute__((vector_size(16)))     int    v4i;

constexpr int NB = 32;          // batches
constexpr int N  = 1024;        // Nq == Nk
constexpr int C  = 128;         // qk channels
constexpr int CM = 256;         // v channels
constexpr int BQ = 128;         // q rows per workgroup (8 waves * 16)
constexpr int BK = 64;          // k positions per iteration
constexpr int NIT = N / BK;     // 16

constexpr int KSTR = C + 8;     // sK row stride (bf16): 272B, 16B-aligned, bank-spread
constexpr int VSTR = BK + 8;    // sVT row stride (bf16): 144B

constexpr unsigned KBUF = (unsigned)(BK * KSTR * 2);   // 17408 B per K buffer
constexpr unsigned VBUF = (unsigned)(CM * VSTR * 2);   // 36864 B per V buffer
constexpr unsigned SK_OFF = 0;
constexpr unsigned SV_OFF = 2 * KBUF;                  // 34816
constexpr unsigned SP_OFF = SV_OFF + 2 * VBUF;         // 108544
constexpr unsigned SMEM   = SP_OFF + (unsigned)(8 * 16 * BK * 2); // 124928 B

constexpr size_t KB_BYTES = (size_t)NB * N * C * 2;    // 8 MiB  bf16 K
constexpr size_t VT_BYTES = (size_t)NB * CM * N * 2;   // 16 MiB bf16 V^T

#if defined(__AMDGCN__) && __has_builtin(__builtin_amdgcn_global_load_async_to_lds_b128)
#define HAVE_ASYNC_LDS 1
#else
#define HAVE_ASYNC_LDS 0
#endif

#if HAVE_ASYNC_LDS
typedef __attribute__((address_space(1))) v4i* gv4i_p;  // global
typedef __attribute__((address_space(3))) v4i* lv4i_p;  // LDS
#endif

__device__ inline void cp16B(const __bf16* src, __bf16* dst) {
#if HAVE_ASYNC_LDS
  __builtin_amdgcn_global_load_async_to_lds_b128(
      (gv4i_p)(src), (lv4i_p)(dst), 0, 0);
#else
  *(v8bf*)dst = *(const v8bf*)src;   // synchronous fallback
#endif
}
__device__ inline void wait_async_prev() {   // previous tile's 12 ops retired
#if HAVE_ASYNC_LDS
  asm volatile("s_wait_asynccnt 12" ::: "memory");
#endif
}
__device__ inline void wait_async_all() {
#if HAVE_ASYNC_LDS
  asm volatile("s_wait_asynccnt 0" ::: "memory");
#endif
}

__device__ inline v16bf ld16(const __bf16* p0, const __bf16* p1) {
  v8bf a = *(const v8bf*)p0;   // 16B-aligned ds_load_b128
  v8bf b = *(const v8bf*)p1;
  v16bf r;
#pragma unroll
  for (int e = 0; e < 8; ++e) { r[e] = a[e]; r[e + 8] = b[e]; }
  return r;
}

// Issue one K/V tile (12 async b128 per wave: 4 for K, 8 for V^T).
__device__ inline void issue_tile(const __bf16* Kt, const __bf16* Vt,
                                  __bf16* sKb, __bf16* sVb, int wave, int lane) {
#pragma unroll
  for (int j = 0; j < 4; ++j) {              // K: 64 rows x 256B, contiguous
    const int i   = wave * 4 + j;            // 512B chunk id
    const int row = 2 * i + (lane >> 4);
    const int col = (lane & 15) * 8;         // bf16 elems
    cp16B(Kt + i * 256 + lane * 8, sKb + row * KSTR + col);
  }
#pragma unroll
  for (int j = 0; j < 8; ++j) {              // V^T: 256 rows x 128B
    const int i   = wave * 8 + j;            // 4 m-rows per chunk
    const int m   = 4 * i + (lane >> 3);
    const int col = (lane & 7) * 8;
    cp16B(Vt + (size_t)m * N + col, sVb + m * VSTR + col);
  }
}

// ---------------- pass 1: K f32 -> bf16 (row-major, unchanged layout)
__global__ __launch_bounds__(256)
void cvtK_kernel(const float* __restrict__ K, __bf16* __restrict__ Kb) {
  const size_t t = ((size_t)blockIdx.x * 256 + threadIdx.x) * 4;
  float4 x = *(const float4*)(K + t);
  v4bf y = {(__bf16)x.x, (__bf16)x.y, (__bf16)x.z, (__bf16)x.w};
  *(v4bf*)(Kb + t) = y;
}

// ---------------- pass 2: V f32 [B][N][CM] -> bf16 V^T [B][CM][N]
__global__ __launch_bounds__(256)
void tspV_kernel(const float* __restrict__ V, __bf16* __restrict__ Vt) {
  __shared__ __bf16 tile[64][72];            // [m][k], padded
  const int tid = threadIdx.x;
  const int bid = blockIdx.x;
  const int b   = bid >> 6;                  // 64 tiles per batch
  const int kt  = (bid & 63) >> 2;           // 16 k-tiles
  const int mt  = bid & 3;                   // 4 m-tiles
  const int k0  = kt * 64, m0 = mt * 64;

  {                                          // 4x4 register transpose, coalesced reads
    const int mb4 = (tid & 15) * 4;
    const int kb4 = (tid >> 4) * 4;
    const float* s0 = V + ((size_t)b * N + k0 + kb4) * CM + m0 + mb4;
    float4 r0 = *(const float4*)(s0);
    float4 r1 = *(const float4*)(s0 + CM);
    float4 r2 = *(const float4*)(s0 + 2 * CM);
    float4 r3 = *(const float4*)(s0 + 3 * CM);
    v4bf c0 = {(__bf16)r0.x, (__bf16)r1.x, (__bf16)r2.x, (__bf16)r3.x};
    v4bf c1 = {(__bf16)r0.y, (__bf16)r1.y, (__bf16)r2.y, (__bf16)r3.y};
    v4bf c2 = {(__bf16)r0.z, (__bf16)r1.z, (__bf16)r2.z, (__bf16)r3.z};
    v4bf c3 = {(__bf16)r0.w, (__bf16)r1.w, (__bf16)r2.w, (__bf16)r3.w};
    *(v4bf*)&tile[mb4 + 0][kb4] = c0;
    *(v4bf*)&tile[mb4 + 1][kb4] = c1;
    *(v4bf*)&tile[mb4 + 2][kb4] = c2;
    *(v4bf*)&tile[mb4 + 3][kb4] = c3;
  }
  __syncthreads();
  {                                          // coalesced 32B writes along k
    const int m  = tid >> 2;
    const int q4 = (tid & 3) * 16;
    v8bf a = *(const v8bf*)&tile[m][q4];
    v8bf bb = *(const v8bf*)&tile[m][q4 + 8];
    __bf16* d = Vt + ((size_t)b * CM + m0 + m) * N + k0 + q4;
    *(v8bf*)(d)     = a;
    *(v8bf*)(d + 8) = bb;
  }
}

// ---------------- pass 3: flash attention
__global__ __launch_bounds__(256)
void fa_causal_kernel(const float* __restrict__ Qg, const __bf16* __restrict__ Kb,
                      const __bf16* __restrict__ Vt, float* __restrict__ Og) {
  extern __shared__ __align__(16) char smem[];
  __bf16* sK = (__bf16*)(smem + SK_OFF);   // 2 x [BK][KSTR]
  __bf16* sV = (__bf16*)(smem + SV_OFF);   // 2 x [CM][VSTR]
  __bf16* sP = (__bf16*)(smem + SP_OFF);   // [8 waves][16 q][BK]

  const int tid  = threadIdx.x;
  const int wave = tid >> 5;
  const int lane = tid & 31;
  const int r    = lane & 15;
  const int g    = lane >> 4;

  const int bidx  = blockIdx.x;
  const int batch = bidx >> 3;
  const int qb    = (bidx & 7) * BQ;
  const int qw    = qb + wave * 16;

  const float*  Qbp = Qg + (size_t)batch * N * C;
  const __bf16* Kbp = Kb + (size_t)batch * N * C;
  const __bf16* Vbp = Vt + (size_t)batch * CM * N;
  float*        Obp = Og + (size_t)batch * N * CM;

  // ---- Q tile -> WMMA A-layout bf16 registers (4 chunks of K=32)
  v16bf qA[4];
  {
    const float* qrow = Qbp + (size_t)(qw + r) * C;
#pragma unroll
    for (int cc = 0; cc < 4; ++cc) {
      const float* h0 = qrow + cc * 32 + g * 8;
      const float* h1 = qrow + cc * 32 + 16 + g * 8;
#pragma unroll
      for (int e = 0; e < 8; ++e) {
        qA[cc][e]     = (__bf16)h0[e];
        qA[cc][e + 8] = (__bf16)h1[e];
      }
    }
  }

  float mr[8], dr[8];
#pragma unroll
  for (int v = 0; v < 8; ++v) { mr[v] = -3.0e38f; dr[v] = 0.0f; }
  v8f acc[16];
#pragma unroll
  for (int mt = 0; mt < 16; ++mt)
#pragma unroll
    for (int e = 0; e < 8; ++e) acc[mt][e] = 0.0f;

  // prologue: async-stage tile 0 into buffer 0
  issue_tile(Kbp, Vbp, sK, sV, wave, lane);

  for (int it = 0; it < NIT; ++it) {
    const int  kb  = it * BK;
    const int  cur = it & 1;
    const bool pvq = (kb < qb + BQ);

    // issue next tile into the other buffer (safe: barrier at end of prev iter)
    if (it + 1 < NIT) {
      const int kbn = kb + BK;
      issue_tile(Kbp + (size_t)kbn * C, Vbp + kbn,
                 sK + (1 - cur) * (KBUF / 2), sV + (1 - cur) * (VBUF / 2),
                 wave, lane);
      wait_async_prev();       // tile `it` (12 older ops) has landed
    } else {
      wait_async_all();
    }
    __syncthreads();           // every wave's share of tile `it` visible

    const __bf16* cK = sK + cur * (KBUF / 2);
    const __bf16* cV = sV + cur * (VBUF / 2);

    // ---- S = Q K^T : 4 column tiles x 4 channel chunks
    v8f s[4];
#pragma unroll
    for (int ct = 0; ct < 4; ++ct)
#pragma unroll
      for (int e = 0; e < 8; ++e) s[ct][e] = 0.0f;
#pragma unroll
    for (int cc = 0; cc < 4; ++cc) {
#pragma unroll
      for (int ct = 0; ct < 4; ++ct) {
        const __bf16* kp = cK + (ct * 16 + r) * KSTR + cc * 32 + g * 8;
        v16bf kB = ld16(kp, kp + 16);
        s[ct] = __builtin_amdgcn_wmma_f32_16x16x32_bf16(
            false, qA[cc], false, kB, (short)0, s[ct], false, false);
      }
    }

    // ---- online softmax, denominator over ALL k (full-row softmax)
    float rmax[8], rsum[8], sc[8];
#pragma unroll
    for (int v = 0; v < 8; ++v)
      rmax[v] = fmaxf(fmaxf(s[0][v], s[1][v]), fmaxf(s[2][v], s[3][v]));
#pragma unroll
    for (int off = 1; off <= 8; off <<= 1)
#pragma unroll
      for (int v = 0; v < 8; ++v)
        rmax[v] = fmaxf(rmax[v], __shfl_xor(rmax[v], off, 32));
#pragma unroll
    for (int v = 0; v < 8; ++v) {
      float nm = fmaxf(mr[v], rmax[v]);
      sc[v]   = __expf(mr[v] - nm);
      mr[v]   = nm;
      dr[v]  *= sc[v];
      rsum[v] = 0.0f;
    }
#pragma unroll
    for (int ct = 0; ct < 4; ++ct)
#pragma unroll
      for (int v = 0; v < 8; ++v) {
        float p = __expf(s[ct][v] - mr[v]);
        s[ct][v] = p;
        rsum[v] += p;
      }
#pragma unroll
    for (int off = 1; off <= 8; off <<= 1)
#pragma unroll
      for (int v = 0; v < 8; ++v)
        rsum[v] += __shfl_xor(rsum[v], off, 32);
#pragma unroll
    for (int v = 0; v < 8; ++v) dr[v] += rsum[v];

#pragma unroll
    for (int mt = 0; mt < 16; ++mt)
#pragma unroll
      for (int v = 0; v < 8; ++v) acc[mt][v] *= sc[v];

    if (pvq) {
      // strict-causal-masked probs -> per-wave LDS (C/D layout scatter)
      __bf16* pw = sP + wave * (16 * BK);
#pragma unroll
      for (int ct = 0; ct < 4; ++ct)
#pragma unroll
        for (int v = 0; v < 8; ++v) {
          const int kg = kb + ct * 16 + r;
          const int qg = qw + v + 8 * g;
          float p = (kg < qg) ? s[ct][v] : 0.0f;
          pw[(v + 8 * g) * BK + ct * 16 + r] = (__bf16)p;
        }

      // O += P V : 2 kpos chunks of 32, 16 output m-tiles
#pragma unroll
      for (int pc = 0; pc < 2; ++pc) {
        const __bf16* pr = pw + r * BK + pc * 32 + g * 8;
        v16bf pA = ld16(pr, pr + 16);
#pragma unroll
        for (int mt = 0; mt < 16; ++mt) {
          const __bf16* vp = cV + (mt * 16 + r) * VSTR + pc * 32 + g * 8;
          v16bf vB = ld16(vp, vp + 16);
          acc[mt] = __builtin_amdgcn_wmma_f32_16x16x32_bf16(
              false, pA, false, vB, (short)0, acc[mt], false, false);
        }
      }
    }
    __syncthreads();           // all reads of buffer `cur` done -> reusable
  }

  // ---- epilogue
  float inv[8];
#pragma unroll
  for (int v = 0; v < 8; ++v) inv[v] = 1.0f / dr[v];
#pragma unroll
  for (int mt = 0; mt < 16; ++mt)
#pragma unroll
    for (int v = 0; v < 8; ++v)
      Obp[(size_t)(qw + v + 8 * g) * CM + mt * 16 + r] = acc[mt][v] * inv[v];
}

extern "C" void kernel_launch(void* const* d_in, const int* in_sizes, int n_in,
                              void* d_out, int out_size, void* d_ws, size_t ws_size,
                              hipStream_t stream) {
  (void)in_sizes; (void)n_in; (void)out_size; (void)ws_size;
  const float* q = (const float*)d_in[0];
  const float* k = (const float*)d_in[1];
  const float* v = (const float*)d_in[2];
  float* o = (float*)d_out;
  __bf16* wsK = (__bf16*)d_ws;                           // 8 MiB
  __bf16* wsV = (__bf16*)((char*)d_ws + KB_BYTES);       // 16 MiB

  cvtK_kernel<<<dim3((unsigned)(KB_BYTES / 2 / 4 / 256)), dim3(256), 0, stream>>>(k, wsK);
  tspV_kernel<<<dim3(NB * 64), dim3(256), 0, stream>>>(v, wsV);
  fa_causal_kernel<<<dim3(NB * (N / BQ)), dim3(256), (size_t)SMEM, stream>>>(q, wsK, wsV, o);
}
